// Torch_Ops_Aten_QuantizedLstm_InputLegacyModule_66236985639609
// MI455X (gfx1250) — compile-verified, matrix operand order checked by reading, not run on hardware
//
#include <hip/hip_runtime.h>

// ---------------------------------------------------------------------------
// Bidirectional 2-layer LSTM with fake-quantized weights, CDNA5 (gfx1250).
// B=32, T=256, I=1024, H=512, L=2, dirs=2.  f16 WMMA, f32 accumulate.
// ---------------------------------------------------------------------------

typedef __attribute__((ext_vector_type(16))) _Float16 v16h;
typedef __attribute__((ext_vector_type(8)))  _Float16 v8h;
typedef __attribute__((ext_vector_type(8)))  float    v8f;

#define BATCH 32
#define TT    256
#define II    1024
#define HH    512
#define G4    2048      // 4*H
#define ROWS  8192      // B*T
#define NWG   8         // workgroups per direction in scan
#define PHH   520       // padded LDS row stride (scan), halves
#define PKG   264       // padded LDS row stride (gemm, 256-K phase), halves

// ---- workspace layout (bytes) ----
#define OFF_META  ((size_t)0)                 // amax[8] f32 @0, barrier ctr[4] u32 @128
#define OFF_WIH   ((size_t)256)                               // 4*2048*1024 f16
#define OFF_WHH   (OFF_WIH + (size_t)4*G4*II*2)               // 4*2048*512  f16
#define OFF_BIAS  (OFF_WHH + (size_t)4*G4*HH*2)               // 4*2048 f32
#define OFF_XH    (OFF_BIAS + (size_t)4*G4*4)                 // 8192*1024 f16
#define OFF_YS0   (OFF_XH  + (size_t)ROWS*II*2)               // 8192*1024 f16
#define OFF_HG    (OFF_YS0 + (size_t)ROWS*II*2)               // 2 dirs * 2 parity * 32*512 f16
#define OFF_GP    (OFF_HG  + (size_t)2*2*BATCH*HH*2)          // 2 * 8192*2048 f32

// ---------------------------------------------------------------------------
// CDNA5 async memory -> LDS copy (ASYNCcnt-tracked, no VGPR round trip).
// LDS offset = low 32 bits of the generic pointer (ISA 10.2).
// ---------------------------------------------------------------------------
__device__ __forceinline__ void async_copy_b128(const void* gsrc, void* ldst) {
  asm volatile("global_load_async_to_lds_b128 %0, %1, off"
               :
               : "v"((unsigned)(size_t)ldst), "v"(gsrc)
               : "memory");
}
__device__ __forceinline__ void wait_asynccnt0() {
  asm volatile("s_wait_asynccnt 0" ::: "memory");
}

// ---------------------------------------------------------------------------
// WMMA helpers
// ---------------------------------------------------------------------------
__device__ __forceinline__ v8f wmma_f16(v16h a, v16h b, v8f c) {
  return __builtin_amdgcn_wmma_f32_16x16x32_f16(false, a, false, b, (short)0, c,
                                                false, false);
}

// Load a 16x32 (A: MxK) or 32x16-as-NxK (B from row-major W[n][k]) fragment.
// Per ISA 16-bit layout: lane L -> row r=L&15, half-sel h=L>>4; the 16 halves
// per lane cover K = [8h, 8h+8) U [16+8h, 16+8h+8)  => two b128 loads.
__device__ __forceinline__ v16h load_frag16(const _Float16* base, int ld,
                                            int row0, int k0) {
  const int lane = threadIdx.x & 31;
  const int r = lane & 15;
  const int hs = lane >> 4;
  const _Float16* p = base + (size_t)(row0 + r) * ld + (k0 + 8 * hs);
  v8h lo = *(const v8h*)(p);
  v8h hi = *(const v8h*)(p + 16);
  v16h f;
#pragma unroll
  for (int i = 0; i < 8; ++i) { f[i] = lo[i]; f[i + 8] = hi[i]; }
  return f;
}

__device__ __forceinline__ float sigf(float x) {
  return 1.0f / (1.0f + __expf(-x));
}

// ---------------------------------------------------------------------------
// K0: zero absmax slots + grid-barrier counters (must happen every call)
// ---------------------------------------------------------------------------
__global__ void k_init_meta(unsigned* meta) {
  if (threadIdx.x < 8)  meta[threadIdx.x] = 0u;        // amax as bits (>=0)
  if (threadIdx.x < 4)  meta[32 + threadIdx.x] = 0u;   // barrier counters
}

// ---------------------------------------------------------------------------
// K1: per-tensor absmax over the 8 weight tensors (4 w_ih + 4 w_hh)
// ---------------------------------------------------------------------------
__global__ __launch_bounds__(256) void k_absmax(const float* __restrict__ wih,
                                                const float* __restrict__ whh,
                                                unsigned* __restrict__ amax_u) {
  const int tensor = blockIdx.y;
  const float* base;
  size_t n;
  if (tensor < 4) { base = wih + (size_t)tensor * G4 * II; n = (size_t)G4 * II; }
  else            { base = whh + (size_t)(tensor - 4) * G4 * HH; n = (size_t)G4 * HH; }
  float m = 0.0f;
  for (size_t i = (size_t)blockIdx.x * blockDim.x + threadIdx.x; i < n;
       i += (size_t)gridDim.x * blockDim.x)
    m = fmaxf(m, fabsf(base[i]));
  __shared__ float red[256];
  red[threadIdx.x] = m;
  __syncthreads();
  for (int off = 128; off > 0; off >>= 1) {
    if ((int)threadIdx.x < off)
      red[threadIdx.x] = fmaxf(red[threadIdx.x], red[threadIdx.x + off]);
    __syncthreads();
  }
  if (threadIdx.x == 0) atomicMax(&amax_u[tensor], __float_as_uint(red[0]));
}

// ---------------------------------------------------------------------------
// K2: fake-quantize (per-tensor int8 grid) + convert to f16
// ---------------------------------------------------------------------------
__global__ __launch_bounds__(256) void k_quant(const float* __restrict__ wih,
                                               const float* __restrict__ whh,
                                               const float* __restrict__ amax,
                                               _Float16* __restrict__ wih_h,
                                               _Float16* __restrict__ whh_h) {
  const size_t NIH = (size_t)4 * G4 * II;  // 8388608
  size_t idx = (size_t)blockIdx.x * 256 + threadIdx.x;
  if (idx < NIH) {
    int tensor = (int)(idx >> 21);  // 2048*1024 per tensor
    float s = fmaxf(amax[tensor] * (1.0f / 127.0f), 1e-30f);
    float w = wih[idx];
    float q = fminf(fmaxf(rintf(w / s), -128.0f), 127.0f);
    wih_h[idx] = (_Float16)(q * s);
  } else {
    size_t j = idx - NIH;
    int tensor = 4 + (int)(j >> 20);  // 2048*512 per tensor
    float s = fmaxf(amax[tensor] * (1.0f / 127.0f), 1e-30f);
    float w = whh[j];
    float q = fminf(fmaxf(rintf(w / s), -128.0f), 127.0f);
    whh_h[j] = (_Float16)(q * s);
  }
}

__global__ void k_bias(const float* __restrict__ bih, const float* __restrict__ bhh,
                       float* __restrict__ bias) {
  int i = blockIdx.x * 256 + threadIdx.x;
  if (i < 4 * G4) bias[i] = bih[i] + bhh[i];
}

__global__ void k_tofp16(const float* __restrict__ src, _Float16* __restrict__ dst,
                         int n) {
  int i = blockIdx.x * 256 + threadIdx.x;
  if (i < n) dst[i] = (_Float16)src[i];
}

// ---------------------------------------------------------------------------
// K3: input projection GEMM  C[8192,2048] = A[8192,1024] * W[2048,1024]^T
// Block tile 128m x 256n, 8 waves, wave tile 64m x 64n (4x4 16x16 tiles,
// 16 accumulators -> every fragment feeds 4 WMMAs).  K in 4 phases of 256:
// per phase both A (68 KB) and W (135 KB) strips are async-staged into LDS
// (padded rows, conflict-free b128 reads); the inner loop is LDS-only and
// register double-buffered -> no global-latency stalls in the hot loop.
// ---------------------------------------------------------------------------
__global__ __launch_bounds__(256)
__attribute__((amdgpu_waves_per_eu(2)))
void k_gemm_xw(const _Float16* __restrict__ A, const _Float16* __restrict__ W,
               float* __restrict__ C) {
  const int wave = threadIdx.x >> 5;
  const int lane = threadIdx.x & 31;
  const int r = lane & 15, hs = lane >> 4;
  const int mrow0 = (blockIdx.x >> 3) * 128;   // 64 m-blocks of 128
  const int nblk  = (blockIdx.x & 7) * 256;    // 8 n-blocks of 256
  const int wmr = (wave >> 2) * 64;            // wave m-offset in block (0,64)
  const int wnr = (wave & 3) * 64;             // wave n-offset in block (0..192)

  __shared__ _Float16 a_lds[128 * PKG];        // 67.6 KB
  __shared__ _Float16 w_lds[256 * PKG];        // 135.2 KB

  v8f acc[4][4] = {};

  for (int ph = 0; ph < 4; ++ph) {
    const int kb = ph * 256;
    // async stage A strip: 128 rows x 256 K  (4096 b128 chunks)
#pragma unroll 4
    for (int i = threadIdx.x; i < 128 * 32; i += 256) {
      int row = i >> 5;
      int ck = (i & 31) * 8;
      async_copy_b128(&A[(size_t)(mrow0 + row) * II + kb + ck],
                      &a_lds[row * PKG + ck]);
    }
    // async stage W strip: 256 rows x 256 K  (8192 b128 chunks)
#pragma unroll 4
    for (int i = threadIdx.x; i < 256 * 32; i += 256) {
      int row = i >> 5;
      int ck = (i & 31) * 8;
      async_copy_b128(&W[(size_t)(nblk + row) * II + kb + ck],
                      &w_lds[row * PKG + ck]);
    }
    wait_asynccnt0();
    __syncthreads();

    v16h a0 = load_frag16(a_lds, PKG, wmr + 0, 0);
    v16h a1 = load_frag16(a_lds, PKG, wmr + 16, 0);
    v16h a2 = load_frag16(a_lds, PKG, wmr + 32, 0);
    v16h a3 = load_frag16(a_lds, PKG, wmr + 48, 0);
    v16h b0 = load_frag16(w_lds, PKG, wnr + 0, 0);
    v16h b1 = load_frag16(w_lds, PKG, wnr + 16, 0);
    v16h b2 = load_frag16(w_lds, PKG, wnr + 32, 0);
    v16h b3 = load_frag16(w_lds, PKG, wnr + 48, 0);

#pragma unroll
    for (int kk = 0; kk < 8; ++kk) {
      v16h na0 = a0, na1 = a1, na2 = a2, na3 = a3;
      v16h nb0 = b0, nb1 = b1, nb2 = b2, nb3 = b3;
      if (kk < 7) {
        const int k0 = (kk + 1) * 32;
        na0 = load_frag16(a_lds, PKG, wmr + 0, k0);
        na1 = load_frag16(a_lds, PKG, wmr + 16, k0);
        na2 = load_frag16(a_lds, PKG, wmr + 32, k0);
        na3 = load_frag16(a_lds, PKG, wmr + 48, k0);
        nb0 = load_frag16(w_lds, PKG, wnr + 0, k0);
        nb1 = load_frag16(w_lds, PKG, wnr + 16, k0);
        nb2 = load_frag16(w_lds, PKG, wnr + 32, k0);
        nb3 = load_frag16(w_lds, PKG, wnr + 48, k0);
      }
      acc[0][0] = wmma_f16(a0, b0, acc[0][0]);
      acc[1][0] = wmma_f16(a1, b0, acc[1][0]);
      acc[2][0] = wmma_f16(a2, b0, acc[2][0]);
      acc[3][0] = wmma_f16(a3, b0, acc[3][0]);
      acc[0][1] = wmma_f16(a0, b1, acc[0][1]);
      acc[1][1] = wmma_f16(a1, b1, acc[1][1]);
      acc[2][1] = wmma_f16(a2, b1, acc[2][1]);
      acc[3][1] = wmma_f16(a3, b1, acc[3][1]);
      acc[0][2] = wmma_f16(a0, b2, acc[0][2]);
      acc[1][2] = wmma_f16(a1, b2, acc[1][2]);
      acc[2][2] = wmma_f16(a2, b2, acc[2][2]);
      acc[3][2] = wmma_f16(a3, b2, acc[3][2]);
      acc[0][3] = wmma_f16(a0, b3, acc[0][3]);
      acc[1][3] = wmma_f16(a1, b3, acc[1][3]);
      acc[2][3] = wmma_f16(a2, b3, acc[2][3]);
      acc[3][3] = wmma_f16(a3, b3, acc[3][3]);
      a0 = na0; a1 = na1; a2 = na2; a3 = na3;
      b0 = nb0; b1 = nb1; b2 = nb2; b3 = nb3;
    }
    __syncthreads();
  }

#pragma unroll
  for (int mt = 0; mt < 4; ++mt)
#pragma unroll
    for (int nt = 0; nt < 4; ++nt)
#pragma unroll
      for (int v = 0; v < 8; ++v) {
        int m = mrow0 + wmr + mt * 16 + v + 8 * hs;
        int n = nblk + wnr + nt * 16 + r;
        C[(size_t)m * G4 + n] = acc[mt][nt][v];
      }
}

// ---------------------------------------------------------------------------
// Software grid barrier (monotonic counter) among NWG workgroups of one dir.
// ---------------------------------------------------------------------------
__device__ __forceinline__ void grid_barrier(unsigned* ctr, unsigned target) {
  __syncthreads();
  if (threadIdx.x == 0) {
    __builtin_amdgcn_fence(__ATOMIC_RELEASE, "agent");
    __hip_atomic_fetch_add(ctr, 1u, __ATOMIC_RELAXED, __HIP_MEMORY_SCOPE_AGENT);
    while (__hip_atomic_load(ctr, __ATOMIC_RELAXED, __HIP_MEMORY_SCOPE_AGENT) <
           target) {
      __builtin_amdgcn_s_sleep(2);
    }
    __builtin_amdgcn_fence(__ATOMIC_ACQUIRE, "agent");
  }
  __syncthreads();
}

// ---------------------------------------------------------------------------
// K4: recurrent scan. Grid = 2 dirs * NWG blocks, 128 threads (4 waves),
// one wave per SIMD.  The workgroup's whole w_hh slice (4 gates x 64 rows x
// 512 K = 256 KB) is async-staged into LDS ONCE (weights are time-invariant),
// rows padded to 520 halves for bank-conflict-free b128 fragment reads.
// Per step the K loop is pure LDS: ds_load fragments + 128 WMMAs per wave,
// double-buffered in registers.  h is async-staged to LDS each step; next
// step's gates_pre rows prefetch into registers across the grid barrier;
// c state lives in registers.
// ---------------------------------------------------------------------------
__global__ __launch_bounds__(128)
__attribute__((amdgpu_waves_per_eu(1)))
void k_lstm_scan(
    const float* __restrict__ gates_pre,   // [2][8192][2048]
    const _Float16* __restrict__ whh_all,  // [4][2048][512]
    const float* __restrict__ bias_all,    // [4][2048]
    const float* __restrict__ h0,          // [4][32][512]
    const float* __restrict__ c0,          // [4][32][512]
    _Float16* __restrict__ h_glob,         // [2 dirs][2 parity][32][512]
    _Float16* ys_h,                        // [8192][1024] (layer0) or null
    float* out_f,                          // [8192][1024] (layer1) or null
    float* __restrict__ hn,                // [4][32][512]
    float* __restrict__ cn,                // [4][32][512]
    unsigned* __restrict__ ctr2,           // [2] counters for this layer
    int layer) {
  const int dir = blockIdx.x / NWG;
  const int wg = blockIdx.x % NWG;
  const int wave = threadIdx.x >> 5;
  const int lane = threadIdx.x & 31;
  const int r = lane & 15, hs = lane >> 4;
  const int idx = 2 * layer + dir;
  const int jg = wg * 64 + wave * 16;  // hidden base in [0,512)
  const int jcol = jg + r;
  const bool rev = (dir == 1);

  const float* gp = gates_pre + (size_t)dir * ROWS * G4;
  const _Float16* whh = whh_all + (size_t)idx * G4 * HH;
  const float* bias = bias_all + (size_t)idx * G4;
  _Float16* hgd = h_glob + (size_t)dir * 2 * BATCH * HH;
  unsigned* ctr = ctr2 + dir;

  __shared__ _Float16 whh_lds[256 * PHH];  // 260 KB: WG's w_hh slice, padded
  __shared__ _Float16 hs_lds[BATCH * PHH]; // 32.5 KB staged h for this step

  float breg[4];
#pragma unroll
  for (int q = 0; q < 4; ++q) breg[q] = bias[q * HH + jcol];

  // ---- one-time async stage of the WG's w_hh slice into LDS ----
  // local row rr in [0,256): gate q = rr>>6, global row = q*512 + wg*64 + (rr&63)
#pragma unroll 4
  for (int i = threadIdx.x; i < 256 * 64; i += 128) {
    int rr = i >> 6;
    int ck = (i & 63) * 8;
    int grow = (rr >> 6) * HH + wg * 64 + (rr & 63);
    async_copy_b128(&whh[(size_t)grow * HH + ck], &whh_lds[rr * PHH + ck]);
  }

  // init c registers and parity-0 h buffer from h0/c0
  v8f creg[2];
#pragma unroll
  for (int mt = 0; mt < 2; ++mt)
#pragma unroll
    for (int v = 0; v < 8; ++v) {
      int m = mt * 16 + v + 8 * hs;
      creg[mt][v] = c0[(size_t)idx * BATCH * HH + m * HH + jcol];
      hgd[m * HH + jcol] = (_Float16)h0[(size_t)idx * BATCH * HH + m * HH + jcol];
    }

  // preload gates_pre rows for the first step into registers
  float gpn[4][2][8];
  {
    const int t0 = rev ? (TT - 1) : 0;
#pragma unroll
    for (int q = 0; q < 4; ++q)
#pragma unroll
      for (int mt = 0; mt < 2; ++mt)
#pragma unroll
        for (int v = 0; v < 8; ++v) {
          int m = mt * 16 + v + 8 * hs;
          gpn[q][mt][v] = gp[((size_t)(m * TT + t0)) * G4 + q * HH + jcol];
        }
  }

  wait_asynccnt0();   // w_hh slice resident in LDS
  unsigned phase = 1;
  grid_barrier(ctr, phase * NWG);
  ++phase;

  const int jb = wave * 16;  // wave's local row base inside a 64-row gate block

  for (int s = 0; s < TT; ++s) {
    const int t = rev ? (TT - 1 - s) : s;
    const _Float16* hsrc = hgd + (size_t)(s & 1) * BATCH * HH;
    _Float16* hdst = hgd + (size_t)((s + 1) & 1) * BATCH * HH;

    // accumulators from prefetched input projection + bias (pure VALU)
    v8f acc[4][2];
#pragma unroll
    for (int q = 0; q < 4; ++q)
#pragma unroll
      for (int mt = 0; mt < 2; ++mt)
#pragma unroll
        for (int v = 0; v < 8; ++v)
          acc[q][mt][v] = gpn[q][mt][v] + breg[q];

    // async-stage h_prev (shared across all 4 waves) into padded LDS
#pragma unroll
    for (int i = 0; i < 16; ++i) {
      int e = threadIdx.x + i * 128;   // 2048 chunks of 8 halves
      int rr = e >> 6;
      int ck = (e & 63) * 8;
      async_copy_b128(&hsrc[rr * HH + ck], &hs_lds[rr * PHH + ck]);
    }
    wait_asynccnt0();
    __syncthreads();

    // gates += h_prev @ w_hh^T  (K = 512, 16 pipelined LDS-only steps of 32)
    v16h a0 = load_frag16(hs_lds, PHH, 0, 0);
    v16h a1 = load_frag16(hs_lds, PHH, 16, 0);
    v16h b0 = load_frag16(whh_lds, PHH, 0 * 64 + jb, 0);
    v16h b1 = load_frag16(whh_lds, PHH, 1 * 64 + jb, 0);
    v16h b2 = load_frag16(whh_lds, PHH, 2 * 64 + jb, 0);
    v16h b3 = load_frag16(whh_lds, PHH, 3 * 64 + jb, 0);
#pragma unroll
    for (int kk = 0; kk < 16; ++kk) {
      v16h na0 = a0, na1 = a1, nb0 = b0, nb1 = b1, nb2 = b2, nb3 = b3;
      if (kk < 15) {
        const int k0 = (kk + 1) * 32;
        na0 = load_frag16(hs_lds, PHH, 0, k0);
        na1 = load_frag16(hs_lds, PHH, 16, k0);
        nb0 = load_frag16(whh_lds, PHH, 0 * 64 + jb, k0);
        nb1 = load_frag16(whh_lds, PHH, 1 * 64 + jb, k0);
        nb2 = load_frag16(whh_lds, PHH, 2 * 64 + jb, k0);
        nb3 = load_frag16(whh_lds, PHH, 3 * 64 + jb, k0);
      }
      acc[0][0] = wmma_f16(a0, b0, acc[0][0]);
      acc[0][1] = wmma_f16(a1, b0, acc[0][1]);
      acc[1][0] = wmma_f16(a0, b1, acc[1][0]);
      acc[1][1] = wmma_f16(a1, b1, acc[1][1]);
      acc[2][0] = wmma_f16(a0, b2, acc[2][0]);
      acc[2][1] = wmma_f16(a1, b2, acc[2][1]);
      acc[3][0] = wmma_f16(a0, b3, acc[3][0]);
      acc[3][1] = wmma_f16(a1, b3, acc[3][1]);
      a0 = na0; a1 = na1; b0 = nb0; b1 = nb1; b2 = nb2; b3 = nb3;
    }

    // prefetch next step's gates_pre rows (completes under the barrier;
    // consumed only next iteration)
    if (s + 1 < TT) {
      const int tn = rev ? (TT - 2 - s) : (s + 1);
#pragma unroll
      for (int q = 0; q < 4; ++q)
#pragma unroll
        for (int mt = 0; mt < 2; ++mt)
#pragma unroll
          for (int v = 0; v < 8; ++v) {
            int m = mt * 16 + v + 8 * hs;
            gpn[q][mt][v] = gp[((size_t)(m * TT + tn)) * G4 + q * HH + jcol];
          }
    }

    // activations, state update, outputs
#pragma unroll
    for (int mt = 0; mt < 2; ++mt)
#pragma unroll
      for (int v = 0; v < 8; ++v) {
        int m = mt * 16 + v + 8 * hs;
        float iv = acc[0][mt][v], fv = acc[1][mt][v];
        float gv = acc[2][mt][v], ov = acc[3][mt][v];
        float cne = sigf(fv) * creg[mt][v] + sigf(iv) * tanhf(gv);
        float hne = sigf(ov) * tanhf(cne);
        creg[mt][v] = cne;
        hdst[m * HH + jcol] = (_Float16)hne;
        size_t orow = (size_t)(m * TT + t);
        if (ys_h) ys_h[orow * 1024 + dir * HH + jcol] = (_Float16)hne;
        if (out_f) out_f[orow * 1024 + dir * HH + jcol] = hne;
        if (s == TT - 1) {
          hn[(size_t)idx * BATCH * HH + m * HH + jcol] = hne;
          cn[(size_t)idx * BATCH * HH + m * HH + jcol] = cne;
        }
      }

    grid_barrier(ctr, phase * NWG);
    ++phase;
  }
}

// ---------------------------------------------------------------------------
extern "C" void kernel_launch(void* const* d_in, const int* in_sizes, int n_in,
                              void* d_out, int out_size, void* d_ws,
                              size_t ws_size, hipStream_t stream) {
  (void)in_sizes; (void)n_in; (void)out_size; (void)ws_size;

  const float* x   = (const float*)d_in[0];
  const float* h0  = (const float*)d_in[1];
  const float* c0  = (const float*)d_in[2];
  const float* wih = (const float*)d_in[3];
  const float* whh = (const float*)d_in[4];
  const float* bih = (const float*)d_in[5];
  const float* bhh = (const float*)d_in[6];

  float* out = (float*)d_out;                 // [32][256][1024]
  float* hn = out + (size_t)ROWS * 1024;      // [4][32][512]
  float* cn = hn + (size_t)4 * BATCH * HH;

  char* ws = (char*)d_ws;
  unsigned* meta   = (unsigned*)(ws + OFF_META);
  float* amax      = (float*)(ws + OFF_META);
  unsigned* ctrs   = (unsigned*)(ws + OFF_META + 128);
  _Float16* wih_h  = (_Float16*)(ws + OFF_WIH);
  _Float16* whh_h  = (_Float16*)(ws + OFF_WHH);
  float* bias      = (float*)(ws + OFF_BIAS);
  _Float16* x_h    = (_Float16*)(ws + OFF_XH);
  _Float16* ys0_h  = (_Float16*)(ws + OFF_YS0);
  _Float16* h_glob = (_Float16*)(ws + OFF_HG);
  float* gates     = (float*)(ws + OFF_GP);

  // 0: reset counters/absmax (determinism across graph replays)
  k_init_meta<<<1, 64, 0, stream>>>(meta);

  // 1: per-tensor absmax
  k_absmax<<<dim3(64, 8), 256, 0, stream>>>(wih, whh, meta);

  // 2: fake-quant weights -> f16; bias combine; x -> f16
  k_quant<<<(unsigned)(((size_t)4 * G4 * II + (size_t)4 * G4 * HH) / 256), 256, 0,
            stream>>>(wih, whh, amax, wih_h, whh_h);
  k_bias<<<(4 * G4 + 255) / 256, 256, 0, stream>>>(bih, bhh, bias);
  k_tofp16<<<(ROWS * II + 255) / 256, 256, 0, stream>>>(x, x_h, ROWS * II);

  // ---- layer 0 ----
  for (int d = 0; d < 2; ++d)
    k_gemm_xw<<<512, 256, 0, stream>>>(x_h, wih_h + (size_t)d * G4 * II,
                                       gates + (size_t)d * ROWS * G4);
  k_lstm_scan<<<2 * NWG, 128, 0, stream>>>(gates, whh_h, bias, h0, c0, h_glob,
                                           ys0_h, nullptr, hn, cn, ctrs + 0, 0);

  // ---- layer 1 (input = concat(fwd,bwd) f16, same K=1024) ----
  for (int d = 0; d < 2; ++d)
    k_gemm_xw<<<512, 256, 0, stream>>>(ys0_h, wih_h + (size_t)(2 + d) * G4 * II,
                                       gates + (size_t)d * ROWS * G4);
  k_lstm_scan<<<2 * NWG, 128, 0, stream>>>(gates, whh_h, bias, h0, c0, h_glob,
                                           nullptr, out, hn, cn, ctrs + 2, 1);
}
